// RNNLM_45878840656132
// MI455X (gfx1250) — compile-verified, matrix operand order checked by reading, not run on hardware
//
#include <hip/hip_runtime.h>
#include <hip/hip_bf16.h>
#include <math.h>

typedef __attribute__((ext_vector_type(16))) __bf16 v16bf;
typedef __attribute__((ext_vector_type(8)))  float  v8f;
typedef __attribute__((ext_vector_type(4)))  unsigned int u32x4;
typedef __attribute__((ext_vector_type(8)))  int i32x8;
typedef __attribute__((ext_vector_type(4)))  int i32x4;

#define LLEN 128
#define BB   32
#define VV   50257
#define EE   512
#define HH   512
#define TT   127            // L-1 steps
#define ROWS 4064           // TT*BB
#define GATE 2048           // 4*H
#define KCAT 1536           // E + H + H
#define NPAD 50272          // ceil(V/16)*16
#define NTILES 3142         // NPAD/16
#define NSPLIT 8
#define CHUNK 393           // ceil(NTILES/NSPLIT)
#define LSTRIDE 520         // LDS panel row stride in elements (512 + 4-DWORD TDM pad)

static __device__ __forceinline__ unsigned short f2bf(float f) {
  union { float f; unsigned u; } v; v.f = f;
  unsigned lsb = (v.u >> 16) & 1u;
  return (unsigned short)((v.u + 0x7fffu + lsb) >> 16);
}
static __device__ __forceinline__ float bf2f(unsigned short h) {
  union { unsigned u; float f; } v; v.u = ((unsigned)h) << 16;
  return v.f;
}

// ---- TDM: stage one B panel (16 vocab cols x 512 K, bf16) into LDS ----
// D# per CDNA5 ISA: 2D tensor, data_size=2B,
// tile_dim0=512 (K), tile_dim1=16 (cols), row stride 512,
// LDS padding: every 256 DWORDs (one row) add 4 DWORDs -> row stride 520 elems.
static __device__ __forceinline__ void tdm_load_panel(const unsigned short* gsrc,
                                                      unsigned short* lbuf) {
#if __has_builtin(__builtin_amdgcn_tensor_load_to_lds)
  unsigned long long ga = (unsigned long long)(size_t)gsrc;
  unsigned lds_off = (unsigned)(size_t)lbuf;   // generic shared ptr: low 32 bits = LDS offset
  u32x4 g0;
  g0[0] = 1u;                                        // count=1, user descriptor
  g0[1] = lds_off;                                   // lds_addr (bytes)
  g0[2] = (unsigned)(ga & 0xFFFFFFFFu);              // global_addr[31:0]
  g0[3] = (unsigned)((ga >> 32) & 0x1FFFFFFu) | (2u << 30);  // global_addr[56:32] | type=2
  i32x8 g1;
  g1[0] = (1 << 16) | (1 << 20) | (7 << 22) | (3 << 25); // data_size=2B, pad_en, interval=256dw, amount=4dw
  g1[1] = (int)(512u << 16);        // tensor_dim0 = 512 (low16 at bits 63:48)
  g1[2] = (int)((unsigned)NPAD << 16); // tensor_dim1 low16 at bits 95:80
  g1[3] = (int)(512u << 16);        // tile_dim0 = 512 at bits 127:112
  g1[4] = 16;                       // tile_dim1 = 16, tile_dim2 = 0
  g1[5] = 512;                      // tensor_dim0_stride low32
  g1[6] = 0;                        // stride hi / tensor_dim1_stride
  g1[7] = 0;
  i32x4 z = {0, 0, 0, 0};
#if __has_include(<hip/amd_detail/amd_gfx1250_TDM.h>)
  i32x8 z8 = {};
  __builtin_amdgcn_tensor_load_to_lds(g0, g1, z, z, z8, 0);
#else
  __builtin_amdgcn_tensor_load_to_lds(g0, g1, z, z, 0);
#endif
#else
  // fallback: synchronous copy by the issuing wave
  int lane = threadIdx.x & 31;
  for (int r = 0; r < 16; ++r)
    for (int j = lane; j < 512; j += 32)
      lbuf[r * LSTRIDE + j] = gsrc[r * 512 + j];
#endif
}

static __device__ __forceinline__ void tdm_wait(int pending) {
#if __has_builtin(__builtin_amdgcn_s_wait_tensorcnt)
  if (pending) __builtin_amdgcn_s_wait_tensorcnt(1);
  else         __builtin_amdgcn_s_wait_tensorcnt(0);
#else
  (void)pending;
#endif
}

// ---- weight packing: Wcat = [W_ih | W_hh] bf16, row-major (GATE x KCAT) ----
__global__ void k_build_wcat(const float* __restrict__ Wih, const float* __restrict__ Whh,
                             unsigned short* __restrict__ wcat) {
  size_t idx = (size_t)blockIdx.x * blockDim.x + threadIdx.x;
  if (idx >= (size_t)GATE * KCAT) return;
  int r = (int)(idx / KCAT), k = (int)(idx % KCAT);
  float v = (k < EE + HH) ? Wih[(size_t)r * (EE + HH) + k]
                          : Whh[(size_t)r * HH + (k - EE - HH)];
  wcat[idx] = f2bf(v);
}

__global__ void k_bias(const float* __restrict__ bih, const float* __restrict__ bhh,
                       float* __restrict__ bcat) {
  int i = blockIdx.x * blockDim.x + threadIdx.x;
  if (i < GATE) bcat[i] = bih[i] + bhh[i];
}

__global__ void k_conv_wv(const float* __restrict__ Wv, unsigned short* __restrict__ wvb) {
  size_t idx = (size_t)blockIdx.x * blockDim.x + threadIdx.x;
  if (idx >= (size_t)NPAD * HH) return;
  float v = (idx < (size_t)VV * HH) ? Wv[idx] : 0.0f;
  wvb[idx] = f2bf(v);
}

__global__ void k_embed(const int* __restrict__ sents, const float* __restrict__ Emb,
                        unsigned short* __restrict__ embb) {
  size_t idx = (size_t)blockIdx.x * blockDim.x + threadIdx.x;
  if (idx >= (size_t)ROWS * EE) return;
  int k  = (int)(idx % EE);
  int tb = (int)(idx / EE);
  int tok = sents[tb];
  embb[idx] = f2bf(Emb[(size_t)tok * EE + k]);
}

__global__ void k_init(float* __restrict__ c, unsigned short* __restrict__ hb,
                       unsigned short* __restrict__ ob, float* __restrict__ dout) {
  int i = blockIdx.x * blockDim.x + threadIdx.x;
  if (i < BB * HH) { c[i] = 0.f; hb[i] = 0; ob[i] = 0; }
  if (i < BB) dout[i] = 0.f;
}

// ---- one LSTM step: gates = [x_t | o_prev | h] @ Wcat^T (M=32,N=2048,K=1536) ----
// K loop split into 3 straight-line regions (no per-kt branch chain).
__global__ void k_step_gemm(const unsigned short* __restrict__ embb,
                            const unsigned short* __restrict__ ob,
                            const unsigned short* __restrict__ hb,
                            const unsigned short* __restrict__ wcat,
                            float* __restrict__ gates, int t) {
  const int lane = threadIdx.x & 31;
  const int wv   = threadIdx.x >> 5;
  const int n    = blockIdx.x * 4 + wv;   // N-tile 0..127
  const int lm   = lane & 15;
  const int half = lane >> 4;
  const int col  = n * 16 + lm;
  const int hoff = half * 16;

  const unsigned short* e0 = embb + ((size_t)t * BB + lm) * EE + hoff;
  const unsigned short* e1 = embb + ((size_t)t * BB + 16 + lm) * EE + hoff;
  const unsigned short* o0 = ob + (size_t)lm * HH + hoff;
  const unsigned short* o1 = ob + (size_t)(16 + lm) * HH + hoff;
  const unsigned short* h0 = hb + (size_t)lm * HH + hoff;
  const unsigned short* h1 = hb + (size_t)(16 + lm) * HH + hoff;
  const unsigned short* wb = wcat + (size_t)col * KCAT + hoff;
  __builtin_prefetch(wb, 0, 1);

  v8f acc0 = {}; v8f acc1 = {};
  #pragma unroll 4
  for (int kt = 0; kt < 16; ++kt) {
    v16bf av0 = *(const v16bf*)(e0 + kt * 32);
    v16bf av1 = *(const v16bf*)(e1 + kt * 32);
    v16bf bv  = *(const v16bf*)(wb + kt * 32);
    acc0 = __builtin_amdgcn_wmma_f32_16x16x32_bf16(false, av0, false, bv, (short)0, acc0, false, false);
    acc1 = __builtin_amdgcn_wmma_f32_16x16x32_bf16(false, av1, false, bv, (short)0, acc1, false, false);
  }
  #pragma unroll 4
  for (int kt = 0; kt < 16; ++kt) {
    v16bf av0 = *(const v16bf*)(o0 + kt * 32);
    v16bf av1 = *(const v16bf*)(o1 + kt * 32);
    v16bf bv  = *(const v16bf*)(wb + EE + kt * 32);
    acc0 = __builtin_amdgcn_wmma_f32_16x16x32_bf16(false, av0, false, bv, (short)0, acc0, false, false);
    acc1 = __builtin_amdgcn_wmma_f32_16x16x32_bf16(false, av1, false, bv, (short)0, acc1, false, false);
  }
  #pragma unroll 4
  for (int kt = 0; kt < 16; ++kt) {
    v16bf av0 = *(const v16bf*)(h0 + kt * 32);
    v16bf av1 = *(const v16bf*)(h1 + kt * 32);
    v16bf bv  = *(const v16bf*)(wb + EE + HH + kt * 32);
    acc0 = __builtin_amdgcn_wmma_f32_16x16x32_bf16(false, av0, false, bv, (short)0, acc0, false, false);
    acc1 = __builtin_amdgcn_wmma_f32_16x16x32_bf16(false, av1, false, bv, (short)0, acc1, false, false);
  }
  #pragma unroll
  for (int r = 0; r < 8; ++r) {
    int row0 = r + half * 8;
    gates[(size_t)row0 * GATE + col]        = acc0[r];
    gates[(size_t)(16 + row0) * GATE + col] = acc1[r];
  }
}

__global__ void k_step_elem(const float* __restrict__ gates, const float* __restrict__ bcat,
                            float* __restrict__ c, unsigned short* __restrict__ hb,
                            unsigned short* __restrict__ ob, unsigned short* __restrict__ outsb,
                            int t) {
  int idx = blockIdx.x * blockDim.x + threadIdx.x;
  if (idx >= BB * HH) return;
  int b = idx / HH, j = idx % HH;
  const float* g = gates + (size_t)b * GATE;
  float ig = 1.f / (1.f + expf(-(g[j]          + bcat[j])));
  float fg = 1.f / (1.f + expf(-(g[HH + j]     + bcat[HH + j])));
  float gg = tanhf(        g[2 * HH + j]       + bcat[2 * HH + j]);
  float og = 1.f / (1.f + expf(-(g[3 * HH + j] + bcat[3 * HH + j])));
  float cn = fg * c[idx] + ig * gg;
  c[idx] = cn;
  float hn = og * tanhf(cn);
  hb[idx] = f2bf(hn);
  float outv = tanhf(hn);
  ob[idx] = f2bf(outv);
  outsb[((size_t)t * BB + b) * HH + j] = f2bf(outv);
}

// ---- vocab GEMM, TDM-staged B panels in LDS, fused online logsumexp ----
// grid: 64 M-groups x 8 N-splits. Each WG: 4 waves x 16-row M-tiles sharing
// double-buffered 16-col B panels. Partial (max,sum) per N-split to global.
__global__ void k_vocab_lse(const unsigned short* __restrict__ outsb,
                            const unsigned short* __restrict__ wvb,
                            const float* __restrict__ bvoc,
                            float* __restrict__ pm, float* __restrict__ ps) {
  __shared__ unsigned short sB[2][16 * LSTRIDE];

  const int lane   = threadIdx.x & 31;
  const int wv     = threadIdx.x >> 5;
  const int lm     = lane & 15;
  const int half   = lane >> 4;
  const int mgroup = blockIdx.x >> 3;
  const int nsplit = blockIdx.x & 7;
  const int mtile  = mgroup * 4 + wv;          // 0..255, valid < 254
  const bool valid = mtile < (ROWS / 16);
  const int arow   = valid ? (mtile * 16 + lm) : lm;

  // A fragments resident in registers for the whole N sweep
  v16bf afrag[16];
  #pragma unroll
  for (int kt = 0; kt < 16; ++kt)
    afrag[kt] = *(const v16bf*)(outsb + (size_t)arow * HH + kt * 32 + half * 16);

  float m8[8], s8[8];
  #pragma unroll
  for (int r = 0; r < 8; ++r) { m8[r] = -1e30f; s8[r] = 0.f; }

  const int nbeg = nsplit * CHUNK;
  const int nend = (nbeg + CHUNK < NTILES) ? (nbeg + CHUNK) : NTILES;
  const int npanels = nend - nbeg;

  if (wv == 0) {
    tdm_load_panel(wvb + (size_t)nbeg * 16 * HH, &sB[0][0]);
    if (npanels > 1) tdm_load_panel(wvb + (size_t)(nbeg + 1) * 16 * HH, &sB[1][0]);
  }

  for (int i = 0; i < npanels; ++i) {
    if (wv == 0) tdm_wait(i + 1 < npanels);
    __syncthreads();                       // panel i resident in sB[i&1]

    const unsigned short* bp = &sB[i & 1][0] + (size_t)lm * LSTRIDE + half * 16;
    v8f acc = {};
    #pragma unroll
    for (int kt = 0; kt < 16; ++kt) {
      v16bf bv = *(const v16bf*)(bp + kt * 32);   // ds_load from shared panel
      acc = __builtin_amdgcn_wmma_f32_16x16x32_bf16(false, afrag[kt], false, bv, (short)0, acc, false, false);
    }
    int col = (nbeg + i) * 16 + lm;
    if (col < VV) {
      float bias = bvoc[col];
      #pragma unroll
      for (int r = 0; r < 8; ++r) {
        float v = acc[r] + bias;
        if (v > m8[r]) { s8[r] = s8[r] * expf(m8[r] - v) + 1.f; m8[r] = v; }
        else           { s8[r] += expf(v - m8[r]); }
      }
    }
    __syncthreads();                       // everyone done with sB[i&1]
    if (wv == 0 && i + 2 < npanels)
      tdm_load_panel(wvb + (size_t)(nbeg + i + 2) * 16 * HH, &sB[i & 1][0]);
  }

  // combine across the 16 lanes holding each row (C layout halves)
  #pragma unroll
  for (int mask = 1; mask < 16; mask <<= 1) {
    #pragma unroll
    for (int r = 0; r < 8; ++r) {
      float mo = __shfl_xor(m8[r], mask, 32);
      float so = __shfl_xor(s8[r], mask, 32);
      float mm = fmaxf(m8[r], mo);
      s8[r] = s8[r] * expf(m8[r] - mm) + so * expf(mo - mm);
      m8[r] = mm;
    }
  }
  if (valid && lm == 0) {
    #pragma unroll
    for (int r = 0; r < 8; ++r) {
      int row = mtile * 16 + r + half * 8;
      pm[(size_t)row * NSPLIT + nsplit] = m8[r];
      ps[(size_t)row * NSPLIT + nsplit] = s8[r];
    }
  }
}

__global__ void k_lse_reduce(const float* __restrict__ pm, const float* __restrict__ ps,
                             float* __restrict__ lse) {
  int row = blockIdx.x * blockDim.x + threadIdx.x;
  if (row >= ROWS) return;
  float m = -1e30f, s = 0.f;
  #pragma unroll
  for (int j = 0; j < NSPLIT; ++j) {
    float mo = pm[(size_t)row * NSPLIT + j], so = ps[(size_t)row * NSPLIT + j];
    float mm = fmaxf(m, mo);
    s = s * expf(m - mm) + so * expf(mo - mm);
    m = mm;
  }
  lse[row] = m + logf(s);
}

__global__ void k_score(const int* __restrict__ sents,
                        const unsigned short* __restrict__ outsb,
                        const float* __restrict__ Wv,
                        const float* __restrict__ bvoc,
                        const float* __restrict__ lse,
                        float* __restrict__ dout) {
  int lane = threadIdx.x & 31;
  int wv   = threadIdx.x >> 5;
  int row  = blockIdx.x * 4 + wv;
  if (row >= ROWS) return;
  int t = row >> 5, b = row & 31;
  int tok = sents[(t + 1) * BB + b];
  float dot = 0.f;
  #pragma unroll
  for (int j = 0; j < 16; ++j) {
    int k = lane + j * 32;
    dot += bf2f(outsb[(size_t)row * HH + k]) * Wv[(size_t)tok * HH + k];
  }
  #pragma unroll
  for (int mask = 1; mask < 32; mask <<= 1) dot += __shfl_xor(dot, mask, 32);
  if (lane == 0) {
    float sc = (tok != 0) ? (dot + bvoc[tok] - lse[row]) : 0.f;   // PAD == 0
    atomicAdd(&dout[b], sc);
  }
}

extern "C" void kernel_launch(void* const* d_in, const int* in_sizes, int n_in,
                              void* d_out, int out_size, void* d_ws, size_t ws_size,
                              hipStream_t stream) {
  (void)in_sizes; (void)n_in; (void)out_size; (void)ws_size;
  const int*   sents = (const int*)  d_in[0];
  const float* Emb   = (const float*)d_in[1];
  const float* Wih   = (const float*)d_in[2];
  const float* bih   = (const float*)d_in[3];
  const float* Whh   = (const float*)d_in[4];
  const float* bhh   = (const float*)d_in[5];
  const float* Wv    = (const float*)d_in[8];
  const float* bvoc  = (const float*)d_in[9];
  float* dout = (float*)d_out;

  char* ws = (char*)d_ws;
  size_t off = 0;
  auto alloc = [&](size_t bytes) {
    char* p = ws + off;
    off = (off + bytes + 255) & ~(size_t)255;
    return p;
  };
  unsigned short* wcat  = (unsigned short*)alloc((size_t)GATE * KCAT * 2);
  float*          bcat  = (float*)         alloc((size_t)GATE * 4);
  unsigned short* wvb   = (unsigned short*)alloc((size_t)NPAD * HH * 2);   // 51.5 MB, L2-resident
  unsigned short* embb  = (unsigned short*)alloc((size_t)ROWS * EE * 2);
  unsigned short* outsb = (unsigned short*)alloc((size_t)ROWS * HH * 2);
  unsigned short* hb    = (unsigned short*)alloc((size_t)BB * HH * 2);
  unsigned short* ob    = (unsigned short*)alloc((size_t)BB * HH * 2);
  float*          c     = (float*)         alloc((size_t)BB * HH * 4);
  float*          gates = (float*)         alloc((size_t)BB * GATE * 4);
  float*          pm    = (float*)         alloc((size_t)ROWS * NSPLIT * 4);
  float*          ps    = (float*)         alloc((size_t)ROWS * NSPLIT * 4);
  float*          lseb  = (float*)         alloc((size_t)ROWS * 4);

  k_build_wcat<<<(unsigned)(((size_t)GATE * KCAT + 255) / 256), 256, 0, stream>>>(Wih, Whh, wcat);
  k_bias<<<8, 256, 0, stream>>>(bih, bhh, bcat);
  k_conv_wv<<<(unsigned)(((size_t)NPAD * HH + 255) / 256), 256, 0, stream>>>(Wv, wvb);
  k_embed<<<(unsigned)(((size_t)ROWS * EE + 255) / 256), 256, 0, stream>>>(sents, Emb, embb);
  k_init<<<64, 256, 0, stream>>>(c, hb, ob, dout);

  for (int t = 0; t < TT; ++t) {
    k_step_gemm<<<32, 128, 0, stream>>>(embb, ob, hb, wcat, gates, t);
    k_step_elem<<<64, 256, 0, stream>>>(gates, bcat, c, hb, ob, outsb, t);
  }

  k_vocab_lse<<<64 * NSPLIT, 128, 0, stream>>>(outsb, wvb, bvoc, pm, ps);
  k_lse_reduce<<<(ROWS + 255) / 256, 256, 0, stream>>>(pm, ps, lseb);
  k_score<<<1016, 128, 0, stream>>>(sents, outsb, Wv, bvoc, lseb, dout);
}
